// AttentionHead_77524159693208
// MI455X (gfx1250) — matrix operand order repaired and checked by reading, not compiled, not run on hardware
//
#include <hip/hip_runtime.h>
#include <hip/hip_bf16.h>

// ---------------------------------------------------------------------------
// Fused QKV projection + causal attention (single head, d=1024) for gfx1250.
//   k0: X  f32 -> Xb bf16                    [B*S, D]   (row-major)
//   k1: W  f32 -> Wt bf16 transposed         [3D, D]    ([n][k], frag-friendly)
//   k2: QKV GEMM, LDS-free: direct-global WMMA fragments (L2-resident operands)
//       -> Q[B,S,D] bf16, K[B,S,D] bf16, Vt[B,D,S] bf16
//   k3: flash attention, online softmax; WMMA for Q K^T and P V.
// ---------------------------------------------------------------------------

typedef __attribute__((ext_vector_type(16))) __bf16 v16bf;
typedef __attribute__((ext_vector_type(8)))  __bf16 v8bf;
typedef __attribute__((ext_vector_type(4)))  __bf16 v4bf;
typedef __attribute__((ext_vector_type(8)))  float  v8f;
typedef __attribute__((ext_vector_type(4)))  float  v4f;

static constexpr int BATCH  = 2;
static constexpr int SEQ    = 4096;
static constexpr int DMODEL = 1024;
static constexpr int N3     = 3 * DMODEL;     // 3072
static constexpr int MROWS  = BATCH * SEQ;    // 8192

__device__ __forceinline__ __bf16 f2bf(float f) {
  unsigned u = __builtin_bit_cast(unsigned, f);
  unsigned r = u + 0x7FFFu + ((u >> 16) & 1u);   // round-to-nearest-even
  unsigned short h = (unsigned short)(r >> 16);
  return __builtin_bit_cast(__bf16, h);
}

__device__ __forceinline__ v16bf join16(v8bf a, v8bf b) {
  v16bf r;
#pragma unroll
  for (int i = 0; i < 8; ++i) { r[i] = a[i]; r[i + 8] = b[i]; }
  return r;
}

// ---------------------------------------------------------------------------
// k0: elementwise f32 -> bf16 (8 elements / thread, b128 in / b128 out)
// ---------------------------------------------------------------------------
__global__ __launch_bounds__(256) void cvt_x_kernel(const float* __restrict__ X,
                                                    __bf16* __restrict__ Xb) {
  const size_t i = ((size_t)blockIdx.x * 256 + threadIdx.x) * 8;
  v4f a = *(const v4f*)(X + i);
  v4f b = *(const v4f*)(X + i + 4);
  v8bf o;
#pragma unroll
  for (int e = 0; e < 4; ++e) { o[e] = f2bf(a[e]); o[4 + e] = f2bf(b[e]); }
  *(v8bf*)(Xb + i) = o;
}

// ---------------------------------------------------------------------------
// k1: W[1024][3072] f32 -> Wt[3072][1024] bf16 via 32x32 LDS tile transpose
// ---------------------------------------------------------------------------
__global__ __launch_bounds__(256) void wt_kernel(const float* __restrict__ W,
                                                 __bf16* __restrict__ Wt) {
  __shared__ __align__(16) __bf16 tile[32][33];
  const int n0 = blockIdx.x * 32;
  const int k0 = blockIdx.y * 32;
  const int t  = threadIdx.x;
  const int r  = t >> 3;          // 0..31
  const int c4 = (t & 7) * 4;     // 0,4,...,28
  v4f v = *(const v4f*)(W + (size_t)(k0 + r) * N3 + n0 + c4);
#pragma unroll
  for (int e = 0; e < 4; ++e) tile[c4 + e][r] = f2bf(v[e]);
  __syncthreads();
  v4bf o;
#pragma unroll
  for (int e = 0; e < 4; ++e) o[e] = tile[r][c4 + e];
  *(v4bf*)(Wt + (size_t)(n0 + r) * DMODEL + k0 + c4) = o;
}

// ---------------------------------------------------------------------------
// k2: QKV GEMM, M=8192, N=3072, K=1024. Block = 256 threads (8 waves),
// 128x128 tile; wave (wr,wc) in 4x2 grid owns 32x64 => 2x4 WMMA tiles.
// No LDS: fragments load directly from L2-resident Xb / Wt.
// ---------------------------------------------------------------------------
__global__ __launch_bounds__(256) void qkv_gemm_kernel(
    const __bf16* __restrict__ Xb, const __bf16* __restrict__ Wt,
    __bf16* __restrict__ Qb, __bf16* __restrict__ Kb, __bf16* __restrict__ Vt) {
  const int t    = threadIdx.x;
  const int lane = t & 31;
  const int w    = t >> 5;
  const int wr   = w >> 1;              // 0..3
  const int wc   = w & 1;               // 0..1
  const int m0   = blockIdx.y * 128;
  const int n0   = blockIdx.x * 128;

  const int akb = (lane >> 4) << 4;     // A chunk k-offset (contiguous k, matches B)
  const int bkb = (lane >> 4) << 4;     // B chunk k-offset  (contiguous k, matches A)

  const __bf16* Arow[2];
#pragma unroll
  for (int mt = 0; mt < 2; ++mt)
    Arow[mt] = Xb + (size_t)(m0 + wr * 32 + mt * 16 + (lane & 15)) * DMODEL;
  const __bf16* Brow[4];
#pragma unroll
  for (int nt = 0; nt < 4; ++nt)
    Brow[nt] = Wt + (size_t)(n0 + wc * 64 + nt * 16 + (lane & 15)) * DMODEL;

  v8f acc[2][4];
#pragma unroll
  for (int i = 0; i < 2; ++i)
#pragma unroll
    for (int j = 0; j < 4; ++j) acc[i][j] = (v8f)0.0f;

  for (int kk = 0; kk < DMODEL; kk += 32) {
    if (kk + 64 < DMODEL) {
      // WGP-scope prefetch (locality 3): pull next slabs into all cache levels,
      // shared by the 8 waves of this WGP re-reading the same rows.
      __builtin_prefetch(Arow[0] + kk + 64, 0, 3);
      __builtin_prefetch(Arow[1] + kk + 64, 0, 3);
      __builtin_prefetch(Brow[0] + kk + 64, 0, 3);
      __builtin_prefetch(Brow[1] + kk + 64, 0, 3);
      __builtin_prefetch(Brow[2] + kk + 64, 0, 3);
      __builtin_prefetch(Brow[3] + kk + 64, 0, 3);
    }
    v16bf afrag[2], bfrag[4];
#pragma unroll
    for (int mt = 0; mt < 2; ++mt)
      afrag[mt] = join16(*(const v8bf*)(Arow[mt] + kk + akb),
                         *(const v8bf*)(Arow[mt] + kk + akb + 8));
#pragma unroll
    for (int nt = 0; nt < 4; ++nt)
      bfrag[nt] = join16(*(const v8bf*)(Brow[nt] + kk + bkb),
                         *(const v8bf*)(Brow[nt] + kk + bkb + 8));
#pragma unroll
    for (int mt = 0; mt < 2; ++mt)
#pragma unroll
      for (int nt = 0; nt < 4; ++nt)
        acc[mt][nt] = __builtin_amdgcn_wmma_f32_16x16x32_bf16(
            false, afrag[mt], false, bfrag[nt], (short)0, acc[mt][nt], false, false);
  }

  // ---- epilogue: destination is uniform per block (scalar branch)
  const int sel = n0 >> 10;             // 0 -> Q, 1 -> K, 2 -> V
  if (sel < 2) {
    __bf16* __restrict__ Dst = (sel == 0) ? Qb : Kb;
    const int c0 = (n0 & (DMODEL - 1)) + wc * 64 + (lane & 15);
#pragma unroll
    for (int mt = 0; mt < 2; ++mt)
#pragma unroll
      for (int nt = 0; nt < 4; ++nt) {
        const int col = c0 + nt * 16;
#pragma unroll
        for (int i = 0; i < 8; ++i) {
          const int row = m0 + wr * 32 + mt * 16 + ((lane >> 4) << 3) + i;
          Dst[(size_t)row * DMODEL + col] = f2bf(acc[mt][nt][i]);
        }
      }
  } else {
    // V stored transposed [B][D][S]; 8 acc elements are 8 consecutive s -> b128
    const int d0 = (n0 - 2 * DMODEL) + wc * 64 + (lane & 15);
#pragma unroll
    for (int mt = 0; mt < 2; ++mt) {
      const int s0 = m0 + wr * 32 + mt * 16 + ((lane >> 4) << 3);
      const int bb = s0 >> 12;
      const int ss = s0 & (SEQ - 1);
#pragma unroll
      for (int nt = 0; nt < 4; ++nt) {
        v8bf pk;
#pragma unroll
        for (int i = 0; i < 8; ++i) pk[i] = f2bf(acc[mt][nt][i]);
        *(v8bf*)&Vt[((size_t)bb * DMODEL + d0 + nt * 16) * SEQ + ss] = pk;
      }
    }
  }
}

// ---------------------------------------------------------------------------
// k3: flash attention. Block = 256 threads (8 waves) per 16 query rows.
// Wave w owns d-slice [w*128, w*128+128) of the O accumulator.
// ---------------------------------------------------------------------------
__global__ __launch_bounds__(256) void attn_kernel(
    const __bf16* __restrict__ Qb, const __bf16* __restrict__ Kb,
    const __bf16* __restrict__ Vt, float* __restrict__ out) {
  __shared__ __align__(16) float  sS[8][16][64];   // per-wave partial scores
  __shared__ __align__(16) float  sFin[16][64];    // reduced/masked scores
  __shared__ __align__(16) __bf16 sP[16][64];      // softmax probs (bf16)
  __shared__ float sM[16], sL[16], sAlpha[16];

  const int t     = threadIdx.x;
  const int lane  = t & 31;
  const int w     = t >> 5;
  const int b     = blockIdx.y;
  const int qbase = blockIdx.x * 16;
  const int ds0   = w * 128;

  const __bf16* Q = Qb + (size_t)b * SEQ * DMODEL;
  const __bf16* K = Kb + (size_t)b * SEQ * DMODEL;
  const __bf16* V = Vt + (size_t)b * DMODEL * SEQ;

  // Q fragments for this wave's d-slice: reused for every key tile
  v16bf qf[4];
#pragma unroll
  for (int ks = 0; ks < 4; ++ks) {
    const int r  = qbase + (lane & 15);
    const int kb = ds0 + ks * 32 + ((lane >> 4) << 4);
    qf[ks] = join16(*(const v8bf*)&Q[(size_t)r * DMODEL + kb],
                    *(const v8bf*)&Q[(size_t)r * DMODEL + kb + 8]);
  }

  v8f oacc[8];
#pragma unroll
  for (int i = 0; i < 8; ++i) oacc[i] = (v8f)0.0f;

  if (t < 16) { sM[t] = -1e30f; sL[t] = 0.0f; }
  __syncthreads();

  const int nkt = (qbase + 15) / 64 + 1;   // causal tile bound
  for (int kt = 0; kt < nkt; ++kt) {
    const int kbase = kt * 64;

    // (1) partial scores S_w = Q_slice * K_slice^T  (16x64 f32)
#pragma unroll
    for (int nt = 0; nt < 4; ++nt) {
      v8f sacc = (v8f)0.0f;
#pragma unroll
      for (int ks = 0; ks < 4; ++ks) {
        const int key = kbase + nt * 16 + (lane & 15);
        const int kb  = ds0 + ks * 32 + ((lane >> 4) << 4);
        v16bf kf = join16(*(const v8bf*)&K[(size_t)key * DMODEL + kb],
                          *(const v8bf*)&K[(size_t)key * DMODEL + kb + 8]);
        sacc = __builtin_amdgcn_wmma_f32_16x16x32_bf16(
            false, qf[ks], false, kf, (short)0, sacc, false, false);
      }
#pragma unroll
      for (int i = 0; i < 8; ++i)
        sS[w][((lane >> 4) << 3) + i][nt * 16 + (lane & 15)] = sacc[i];
    }
    __syncthreads();

    // (2) cross-wave reduction + scale + causal mask (4 elems / thread)
    {
      const int e   = t * 4;
      const int row = e >> 6;
      const int c0  = e & 63;
      v4f s = (v4f)0.0f;
#pragma unroll
      for (int ww = 0; ww < 8; ++ww) s += *(const v4f*)&sS[ww][row][c0];
#pragma unroll
      for (int j = 0; j < 4; ++j) {
        float v = s[j] * 0.03125f;                      // 1/sqrt(1024)
        if (kbase + c0 + j > qbase + row) v = -1e30f;   // causal
        sFin[row][c0 + j] = v;
      }
    }
    __syncthreads();

    // (3) online softmax, one thread per row
    if (t < 16) {
      const int row = t;
      const float m_old = sM[row];
      float rmax = -1e30f;
#pragma unroll 8
      for (int c = 0; c < 64; ++c) rmax = fmaxf(rmax, sFin[row][c]);
      const float m_new = fmaxf(m_old, rmax);
      const float alpha = __expf(m_old - m_new);
      float psum = 0.0f;
#pragma unroll 8
      for (int c = 0; c < 64; ++c) {
        const float p = __expf(sFin[row][c] - m_new);
        psum += p;
        sP[row][c] = f2bf(p);
      }
      sM[row]     = m_new;
      sL[row]     = alpha * sL[row] + psum;
      sAlpha[row] = alpha;
    }
    __syncthreads();

    // (4) rescale O, then O += P * V_slice
#pragma unroll
    for (int i = 0; i < 8; ++i) {
      const float a = sAlpha[((lane >> 4) << 3) + i];
#pragma unroll
      for (int nt = 0; nt < 8; ++nt) oacc[nt][i] *= a;
    }
    v16bf pf[2];
#pragma unroll
    for (int ks = 0; ks < 2; ++ks) {
      const int r  = lane & 15;
      const int kb = ks * 32 + ((lane >> 4) << 4);
      pf[ks] = join16(*(const v8bf*)&sP[r][kb], *(const v8bf*)&sP[r][kb + 8]);
    }
#pragma unroll
    for (int nt = 0; nt < 8; ++nt) {
#pragma unroll
      for (int ks = 0; ks < 2; ++ks) {
        const int d  = ds0 + nt * 16 + (lane & 15);
        const int kb = kbase + ks * 32 + ((lane >> 4) << 4);
        v16bf vf = join16(*(const v8bf*)&V[(size_t)d * SEQ + kb],
                          *(const v8bf*)&V[(size_t)d * SEQ + kb + 8]);
        oacc[nt] = __builtin_amdgcn_wmma_f32_16x16x32_bf16(
            false, pf[ks], false, vf, (short)0, oacc[nt], false, false);
      }
    }
    __syncthreads();
  }

  // epilogue: O / l -> out (f32)
#pragma unroll
  for (int i = 0; i < 8; ++i) {
    const int row = ((lane >> 4) << 3) + i;
    const float inv = 1.0f / sL[row];
#pragma unroll
    for (int nt = 0; nt < 8; ++nt) {
      const int d = ds0 + nt * 16 + (lane & 15);
      out[((size_t)b * SEQ + qbase + row) * DMODEL + d] = oacc[nt][i] * inv;
    }
  }
}

// ---------------------------------------------------------------------------
extern "C" void kernel_launch(void* const* d_in, const int* in_sizes, int n_in,
                              void* d_out, int out_size, void* d_ws, size_t ws_size,
                              hipStream_t stream) {
  (void)in_sizes; (void)n_in; (void)out_size; (void)ws_size;
  const float* x = (const float*)d_in[0];
  const float* W = (const float*)d_in[1];
  float* out = (float*)d_out;

  __bf16* Qb = (__bf16*)d_ws;                          // [B,S,D]
  __bf16* Kb = Qb + (size_t)MROWS * DMODEL;            // [B,S,D]
  __bf16* Vt = Kb + (size_t)MROWS * DMODEL;            // [B,D,S]
  __bf16* Xb = Vt + (size_t)MROWS * DMODEL;            // [B*S,D]
  __bf16* Wtp = Xb + (size_t)MROWS * DMODEL;           // [3D,D] (W transposed)

  cvt_x_kernel<<<dim3(MROWS * DMODEL / (256 * 8)), dim3(256), 0, stream>>>(x, Xb);
  wt_kernel<<<dim3(N3 / 32, DMODEL / 32), dim3(256), 0, stream>>>(W, Wtp);
  qkv_gemm_kernel<<<dim3(N3 / 128, MROWS / 128), dim3(256), 0, stream>>>(Xb, Wtp, Qb, Kb, Vt);
  attn_kernel<<<dim3(SEQ / 16, BATCH), dim3(256), 0, stream>>>(Qb, Kb, Vt, out);
}
